// BDB22GNN_3006477107823
// MI455X (gfx1250) — compile-verified, matrix operand order checked by reading, not run on hardware
//
#include <hip/hip_runtime.h>
#include <hip/hip_bf16.h>
#include <math.h>

#define N_NODES 50000
#define N_EDGES 800000
#define N_FEAT  64

typedef __attribute__((ext_vector_type(16))) _Float16 v16h;
typedef __attribute__((ext_vector_type(8)))  float    v8f;

// ---------------------------------------------------------------------------
// Kernel 1: zero workspace (s[50000] followed by t[64])
// ---------------------------------------------------------------------------
__global__ void gnn_init_kernel(float* __restrict__ s) {
    int i = blockIdx.x * 256 + threadIdx.x;
    if (i < N_NODES + N_FEAT) s[i] = 0.0f;
}

// ---------------------------------------------------------------------------
// Kernel 2: s[src[e]] += ew[e]  (800k f32 atomics onto a 200KB L2-resident
// array; dst is provably irrelevant because sum(segment_sum(.)) == sum(.))
// ---------------------------------------------------------------------------
__global__ void gnn_scatter_kernel(const float* __restrict__ ew,
                                   const int*   __restrict__ src,
                                   float* __restrict__ s) {
    int e = blockIdx.x * 256 + threadIdx.x;
    if (e < N_EDGES) {
        __hip_atomic_fetch_add(&s[src[e]], ew[e],
                               __ATOMIC_RELAXED, __HIP_MEMORY_SCOPE_AGENT);
    }
}

// ---------------------------------------------------------------------------
// Kernel 3: t[f] = sum_n s[n] * x[n][f]   (single 12.8MB stream of x)
// Block = 256 threads: lane quad q = feature[4q..4q+3], group g = node slice.
// B128 loads, prefetch 64 rows ahead, LDS reduce, 64 atomics per block.
// ---------------------------------------------------------------------------
#define NODES_PER_BLOCK 512
__global__ void gnn_colsum_kernel(const float* __restrict__ x,
                                  const float* __restrict__ s,
                                  float* __restrict__ t) {
    const int tid = threadIdx.x;      // 0..255
    const int q   = tid & 15;         // feature quad: features 4q..4q+3
    const int g   = tid >> 4;         // node group 0..15
    const int base = blockIdx.x * NODES_PER_BLOCK;

    float4 acc = make_float4(0.f, 0.f, 0.f, 0.f);
    #pragma unroll 4
    for (int i = 0; i < NODES_PER_BLOCK / 16; ++i) {
        int node = base + g + i * 16;
        if (node < N_NODES) {
            // speculative prefetch ~64 rows ahead (gfx1250 global_prefetch_b8)
            __builtin_prefetch(x + (size_t)(node + 64) * N_FEAT + q * 4, 0, 1);
            float  sv = s[node];
            float4 xv = *(const float4*)(x + (size_t)node * N_FEAT + q * 4);
            acc.x += sv * xv.x; acc.y += sv * xv.y;
            acc.z += sv * xv.z; acc.w += sv * xv.w;
        }
    }

    __shared__ float4 red[256];
    red[tid] = acc;
    __syncthreads();

    if (g == 0) {                     // tid 0..15, one per feature quad
        float4 sum = red[q];
        #pragma unroll
        for (int gg = 1; gg < 16; ++gg) {
            float4 r = red[gg * 16 + q];
            sum.x += r.x; sum.y += r.y; sum.z += r.z; sum.w += r.w;
        }
        __hip_atomic_fetch_add(&t[q * 4 + 0], sum.x, __ATOMIC_RELAXED, __HIP_MEMORY_SCOPE_AGENT);
        __hip_atomic_fetch_add(&t[q * 4 + 1], sum.y, __ATOMIC_RELAXED, __HIP_MEMORY_SCOPE_AGENT);
        __hip_atomic_fetch_add(&t[q * 4 + 2], sum.z, __ATOMIC_RELAXED, __HIP_MEMORY_SCOPE_AGENT);
        __hip_atomic_fetch_add(&t[q * 4 + 3], sum.w, __ATOMIC_RELAXED, __HIP_MEMORY_SCOPE_AGENT);
    }
}

// ---------------------------------------------------------------------------
// Kernel 4 (1 wave, 32 threads): pooled = t @ W1 + N*b1 via v_wmma_f32_16x16x32_f16,
// then out = sigmoid(pooled . Wd + bd).
// A-matrix broadcasts t across all 16 M rows (ISA 16-bit A layout: lane m=l%16,
// half=l/16; VGPR v holds K pair kb(v,half)); B tiles of W1 use the identical
// K-striping with N = l%16, so the K pairing is consistent.
// ---------------------------------------------------------------------------
__device__ __forceinline__ int wmma_kb(int v, int half) {
    return (v < 4) ? (2 * v + 8 * half) : (16 + 2 * (v - 4) + 8 * half);
}

__global__ void gnn_head_kernel(const float* __restrict__ t,
                                const float* __restrict__ W1,
                                const float* __restrict__ bias1,
                                const float* __restrict__ Wd,
                                const float* __restrict__ bd,
                                float* __restrict__ out) {
    __shared__ float pooled[N_FEAT];
    const int l    = threadIdx.x;     // 0..31, all lanes active
    const int half = l >> 4;
    const int n    = l & 15;

    // A operands: t broadcast over M rows, two K-chunks (K=0..31, K=32..63)
    v16h a0, a1;
    #pragma unroll
    for (int v = 0; v < 8; ++v) {
        int kb = wmma_kb(v, half);
        a0[2 * v]     = (_Float16)t[kb];
        a0[2 * v + 1] = (_Float16)t[kb + 1];
        a1[2 * v]     = (_Float16)t[32 + kb];
        a1[2 * v + 1] = (_Float16)t[32 + kb + 1];
    }

    #pragma unroll
    for (int nt = 0; nt < 4; ++nt) {  // 4 column tiles of 16
        v16h b0, b1;
        #pragma unroll
        for (int v = 0; v < 8; ++v) {
            int kb = wmma_kb(v, half);
            b0[2 * v]     = (_Float16)W1[(kb)     * N_FEAT + nt * 16 + n];
            b0[2 * v + 1] = (_Float16)W1[(kb + 1) * N_FEAT + nt * 16 + n];
            b1[2 * v]     = (_Float16)W1[(32 + kb)     * N_FEAT + nt * 16 + n];
            b1[2 * v + 1] = (_Float16)W1[(32 + kb + 1) * N_FEAT + nt * 16 + n];
        }
        v8f c = {};
        c = __builtin_amdgcn_wmma_f32_16x16x32_f16(false, a0, false, b0,
                                                   (short)0, c, false, false);
        c = __builtin_amdgcn_wmma_f32_16x16x32_f16(false, a1, false, b1,
                                                   (short)0, c, false, false);
        // C/D layout: VGPR0, lanes 0-15 hold M=0, N=lane
        if (l < 16)
            pooled[nt * 16 + l] = c[0] + (float)N_NODES * bias1[nt * 16 + l];
    }
    __syncthreads();

    if (l == 0) {
        float z = bd[0];
        #pragma unroll 8
        for (int i = 0; i < N_FEAT; ++i) z += pooled[i] * Wd[i];
        out[0] = 1.0f / (1.0f + expf(-z));
    }
}

// ---------------------------------------------------------------------------
// Launcher
// ---------------------------------------------------------------------------
extern "C" void kernel_launch(void* const* d_in, const int* in_sizes, int n_in,
                              void* d_out, int out_size, void* d_ws, size_t ws_size,
                              hipStream_t stream) {
    const float* x     = (const float*)d_in[0];   // [50000,64]
    const float* ew    = (const float*)d_in[1];   // [800000]
    const float* W1    = (const float*)d_in[2];   // [64,64]
    const float* b1    = (const float*)d_in[3];   // [64]
    const float* Wd    = (const float*)d_in[4];   // [64,1]
    const float* bd    = (const float*)d_in[5];   // [1]
    const int*   src   = (const int*)d_in[6];     // [800000]
    // d_in[7] = dst — provably unused after pooling collapse.
    float* out = (float*)d_out;

    float* s = (float*)d_ws;          // [50000]
    float* t = s + N_NODES;           // [64]

    gnn_init_kernel<<<(N_NODES + N_FEAT + 255) / 256, 256, 0, stream>>>(s);
    gnn_scatter_kernel<<<(N_EDGES + 255) / 256, 256, 0, stream>>>(ew, src, s);
    gnn_colsum_kernel<<<(N_NODES + NODES_PER_BLOCK - 1) / NODES_PER_BLOCK, 256, 0, stream>>>(x, s, t);
    gnn_head_kernel<<<1, 32, 0, stream>>>(t, W1, b1, Wd, bd, out);
}
